// MultiHeadAttention_14199161880672
// MI455X (gfx1250) — matrix operand order repaired and checked by reading, not compiled, not run on hardware
//
#include <hip/hip_runtime.h>

typedef __bf16 bf16;
typedef __attribute__((ext_vector_type(8)))  __bf16 v8bf;
typedef __attribute__((ext_vector_type(16))) __bf16 v16bf;
typedef __attribute__((ext_vector_type(8)))  float  v8f;
typedef __attribute__((ext_vector_type(4)))  float  v4f;
typedef __attribute__((ext_vector_type(4)))  int    v4i;

#define D_MODEL 1024
#define NUM_HEADS 16
#define HEAD_DIM 64
#define BATCH 4
#define SEQ 2048
#define M_TOTAL (BATCH * SEQ)   // 8192

// ---------------------------------------------------------------------------
// CDNA5 async global->LDS copy (ASYNCcnt-tracked), guarded so the file
// compiles on toolchains without the builtin (falls back to vector copy).
// Builtin signature (from hipcc diagnostic): (v4i addrspace(1)*,
// v4i addrspace(3)*, imm offset, imm cpol).
// ---------------------------------------------------------------------------
#if defined(__has_builtin)
#if __has_builtin(__builtin_amdgcn_global_load_async_to_lds_b128) && \
    __has_builtin(__builtin_amdgcn_s_wait_asynccnt)
#define USE_ASYNC_LDS 1
#endif
#endif
#ifndef USE_ASYNC_LDS
#define USE_ASYNC_LDS 0
#endif

#if USE_ASYNC_LDS
typedef __attribute__((address_space(1))) v4i gv4i;
typedef __attribute__((address_space(3))) v4i lv4i;
#endif

__device__ __forceinline__ void async_copy_b128(const void* g, void* l) {
#if USE_ASYNC_LDS
    __builtin_amdgcn_global_load_async_to_lds_b128(
        (gv4i*)(g), (lv4i*)(l), 0, 0);
#else
    *(v8bf*)l = *(const v8bf*)g;
#endif
}

__device__ __forceinline__ void async_wait_all() {
#if USE_ASYNC_LDS
    __builtin_amdgcn_s_wait_asynccnt(0);
#endif
}

// ---------------------------------------------------------------------------
// GEMM: C[M,N] = (A[M,K] @ W[K,N] + bias[N]) * outScale
// Block tile 128x128x32, 256 threads (8 waves), wave tile 32x64 (2x4 wmma).
// A staged to LDS as bf16 (async copy when already bf16); W (fp32) converted
// and TRANSPOSED into LDS [n][k] so B-fragments are contiguous per-lane.
// ---------------------------------------------------------------------------
template <bool A_F32, bool OUT_BF16>
__global__ __launch_bounds__(256)
void gemm_bias_kernel(const void* __restrict__ Aptr,
                      const float* __restrict__ W,
                      const float* __restrict__ bias,
                      void* __restrict__ Cptr,
                      int M, int N, int K, float outScale)
{
    constexpr int BM = 128, BN = 128, BK = 32;
    constexpr int LDT = 40;  // padded stride (80B: 16B aligned, conflict-free)

    __shared__ bf16 As[2][BM * LDT];
    __shared__ bf16 Bs[2][BN * LDT];   // [n][k]

    const int tid  = threadIdx.x;
    const int lane = tid & 31;
    const int wave = tid >> 5;
    const int wm   = wave & 3;   // 4 waves along M
    const int wn   = wave >> 2;  // 2 waves along N
    const int half = lane >> 4;
    const int l16  = lane & 15;
    const int bm   = blockIdx.x * BM;
    const int bn   = blockIdx.y * BN;

    v8f acc[2][4];
    for (int mi = 0; mi < 2; ++mi)
        for (int ni = 0; ni < 4; ++ni)
            for (int r = 0; r < 8; ++r) acc[mi][ni][r] = 0.0f;

    auto loadA = [&](int buf, int k0) {
        if (A_F32) {
            const float* A = (const float*)Aptr;
            #pragma unroll
            for (int i = 0; i < 4; ++i) {
                int idx = tid + i * 256;          // 1024 float4 slots
                int row = idx >> 3;               // 8 float4 per 32-wide row
                int c4  = (idx & 7) * 4;
                v4f v = *(const v4f*)(A + (size_t)(bm + row) * K + k0 + c4);
                bf16* dst = &As[buf][row * LDT + c4];
                dst[0] = (bf16)v.x; dst[1] = (bf16)v.y;
                dst[2] = (bf16)v.z; dst[3] = (bf16)v.w;
            }
        } else {
            const bf16* A = (const bf16*)Aptr;
            #pragma unroll
            for (int i = 0; i < 2; ++i) {
                int idx = tid + i * 256;          // 512 v8bf slots
                int row = idx >> 2;               // 4 v8bf per row
                int c8  = (idx & 3) * 8;
                async_copy_b128(A + (size_t)(bm + row) * K + k0 + c8,
                                &As[buf][row * LDT + c8]);
            }
        }
    };

    auto loadB = [&](int buf, int k0) {
        #pragma unroll
        for (int i = 0; i < 4; ++i) {
            int idx = tid + i * 256;              // 1024 float4 slots
            int kr  = idx >> 5;                   // 32 float4 per k-row
            int c4  = (idx & 31) * 4;
            v4f v = *(const v4f*)(W + (size_t)(k0 + kr) * N + bn + c4);
            Bs[buf][(c4 + 0) * LDT + kr] = (bf16)v.x;
            Bs[buf][(c4 + 1) * LDT + kr] = (bf16)v.y;
            Bs[buf][(c4 + 2) * LDT + kr] = (bf16)v.z;
            Bs[buf][(c4 + 3) * LDT + kr] = (bf16)v.w;
        }
    };

    loadA(0, 0);
    loadB(0, 0);
    async_wait_all();
    __syncthreads();

    const int nk = K / BK;
    for (int kt = 0; kt < nk; ++kt) {
        const int buf = kt & 1;
        if (kt + 1 < nk) {
            loadA(buf ^ 1, (kt + 1) * BK);
            loadB(buf ^ 1, (kt + 1) * BK);
        }

        const bf16* Ab = As[buf];
        const bf16* Bb = Bs[buf];

        v16bf afrag[2], bfrag[4];
        #pragma unroll
        for (int mi = 0; mi < 2; ++mi) {
            int row = wm * 32 + mi * 16 + l16;
            int kb  = half * 8;                   // ISA 16-bit A layout
            v8bf lo = *(const v8bf*)&Ab[row * LDT + kb];
            v8bf hi = *(const v8bf*)&Ab[row * LDT + kb + 16];
            #pragma unroll
            for (int j = 0; j < 8; ++j) { afrag[mi][j] = lo[j]; afrag[mi][j + 8] = hi[j]; }
        }
        #pragma unroll
        for (int ni = 0; ni < 4; ++ni) {
            int col = wn * 64 + ni * 16 + l16;
            int kb  = half * 8;                   // ISA 16-bit B layout
            v8bf lo = *(const v8bf*)&Bb[col * LDT + kb];
            v8bf hi = *(const v8bf*)&Bb[col * LDT + kb + 16];
            #pragma unroll
            for (int j = 0; j < 8; ++j) { bfrag[ni][j] = lo[j]; bfrag[ni][j + 8] = hi[j]; }
        }
        #pragma unroll
        for (int mi = 0; mi < 2; ++mi)
            #pragma unroll
            for (int ni = 0; ni < 4; ++ni)
                acc[mi][ni] = __builtin_amdgcn_wmma_f32_16x16x32_bf16(
                    false, afrag[mi], false, bfrag[ni], (short)0, acc[mi][ni],
                    false, false);

        async_wait_all();
        __syncthreads();
    }

    // Epilogue: C layout = VGPR r -> row half*8+r, col = lane%16
    #pragma unroll
    for (int mi = 0; mi < 2; ++mi)
        #pragma unroll
        for (int ni = 0; ni < 4; ++ni)
            #pragma unroll
            for (int r = 0; r < 8; ++r) {
                int row = bm + wm * 32 + mi * 16 + half * 8 + r;
                int col = bn + wn * 64 + ni * 16 + l16;
                float v = (acc[mi][ni][r] + bias[col]) * outScale;
                if (OUT_BF16)
                    ((bf16*)Cptr)[(size_t)row * N + col] = (bf16)v;
                else
                    ((float*)Cptr)[(size_t)row * N + col] = v;
            }
}

// ---------------------------------------------------------------------------
// Flash attention: block = (b, h, 128-query tile); 8 waves x 16 query rows.
// scores = (Q/64) K^T (scale pre-folded into Q projection), online softmax,
// O = P V, ctx written bf16. Q and K tiles staged via async global->LDS.
// ---------------------------------------------------------------------------
__global__ __launch_bounds__(256)
void attn_kernel(const bf16* __restrict__ Qp, const bf16* __restrict__ Kp,
                 const bf16* __restrict__ Vp, bf16* __restrict__ Ctx)
{
    constexpr int QT = 128, KT = 64;
    constexpr int LD = 72;   // pad: 144B row stride, 16B aligned

    __shared__ bf16 Qs[QT * LD];          // [qrow][d]
    __shared__ bf16 Ks[2][KT * LD];       // [key][d]
    __shared__ bf16 Vt[2][HEAD_DIM * LD]; // [d][key] (transposed)
    __shared__ bf16 Ps[8][16 * LD];       // per-wave P tile [row][key]

    const int tid  = threadIdx.x;
    const int lane = tid & 31;
    const int wave = tid >> 5;
    const int half = lane >> 4;
    const int l16  = lane & 15;

    const int b  = blockIdx.z;
    const int h  = blockIdx.y;
    const int q0 = blockIdx.x * QT;

    const size_t rowstride = D_MODEL;
    const bf16* Qg = Qp + ((size_t)b * SEQ + q0) * rowstride + h * HEAD_DIM;

    // Load Q tile (128 x 64) via async copy
    #pragma unroll
    for (int i = 0; i < 4; ++i) {
        int idx = tid + i * 256;          // 1024 v8bf slots
        int row = idx >> 3;
        int c8  = (idx & 7) * 8;
        async_copy_b128(Qg + (size_t)row * rowstride + c8, &Qs[row * LD + c8]);
    }

    auto loadKV = [&](int buf, int k0) {
        const bf16* Kg = Kp + ((size_t)b * SEQ + k0) * rowstride + h * HEAD_DIM;
        const bf16* Vg = Vp + ((size_t)b * SEQ + k0) * rowstride + h * HEAD_DIM;
        #pragma unroll
        for (int i = 0; i < 2; ++i) {
            int idx = tid + i * 256;      // 512 v8bf slots (64x64)
            int row = idx >> 3;
            int c8  = (idx & 7) * 8;
            async_copy_b128(Kg + (size_t)row * rowstride + c8,
                            &Ks[buf][row * LD + c8]);
            v8bf vv = *(const v8bf*)(Vg + (size_t)row * rowstride + c8);
            #pragma unroll
            for (int j = 0; j < 8; ++j)
                Vt[buf][(c8 + j) * LD + row] = vv[j];   // transpose
        }
    };

    loadKV(0, 0);
    async_wait_all();
    __syncthreads();

    // Static Q A-fragments for this wave (rows wave*16 + 0..15), 2 k-steps
    v16bf aq[2];
    #pragma unroll
    for (int ks = 0; ks < 2; ++ks) {
        int row = wave * 16 + l16;
        int kb  = ks * 32 + half * 8;
        v8bf lo = *(const v8bf*)&Qs[row * LD + kb];
        v8bf hi = *(const v8bf*)&Qs[row * LD + kb + 16];
        #pragma unroll
        for (int j = 0; j < 8; ++j) { aq[ks][j] = lo[j]; aq[ks][j + 8] = hi[j]; }
    }

    float m8[8], l8[8];
    v8f Oacc[4];
    #pragma unroll
    for (int r = 0; r < 8; ++r) { m8[r] = -1e30f; l8[r] = 0.0f; }
    #pragma unroll
    for (int ni = 0; ni < 4; ++ni)
        for (int r = 0; r < 8; ++r) Oacc[ni][r] = 0.0f;

    const int ntiles = SEQ / KT;                  // 32

    for (int t = 0; t < ntiles; ++t) {
        const int buf = t & 1;
        if (t + 1 < ntiles) loadKV(buf ^ 1, (t + 1) * KT);

        // ---- S tile = Q K^T (16 x 64 per wave), ks outer so the 4
        // accumulators interleave (avoids same-acc WMMA RAW NOPs) ----
        v8f sc[4];
        #pragma unroll
        for (int kt = 0; kt < 4; ++kt)
            for (int r = 0; r < 8; ++r) sc[kt][r] = 0.0f;

        #pragma unroll
        for (int ks = 0; ks < 2; ++ks) {
            #pragma unroll
            for (int kt = 0; kt < 4; ++kt) {
                int key = kt * 16 + l16;
                int kb  = ks * 32 + half * 8;
                v8bf lo = *(const v8bf*)&Ks[buf][key * LD + kb];
                v8bf hi = *(const v8bf*)&Ks[buf][key * LD + kb + 16];
                v16bf bk;
                #pragma unroll
                for (int j = 0; j < 8; ++j) { bk[j] = lo[j]; bk[j + 8] = hi[j]; }
                sc[kt] = __builtin_amdgcn_wmma_f32_16x16x32_bf16(
                    false, aq[ks], false, bk, (short)0, sc[kt], false, false);
            }
        }

        // ---- online softmax (scores already scaled by 1/64 via Q) ----
        float mnew[8], alpha[8], rsum[8];
        #pragma unroll
        for (int r = 0; r < 8; ++r) {
            float v = -1e30f;
            #pragma unroll
            for (int kt = 0; kt < 4; ++kt)
                v = fmaxf(v, sc[kt][r]);
            // reduce across lanes 0..15 of this half (N striping)
            #pragma unroll
            for (int s = 1; s < 16; s <<= 1)
                v = fmaxf(v, __shfl_xor(v, s, 32));
            mnew[r]  = fmaxf(m8[r], v);
            alpha[r] = __expf(m8[r] - mnew[r]);
            rsum[r]  = 0.0f;
        }

        #pragma unroll
        for (int kt = 0; kt < 4; ++kt) {
            #pragma unroll
            for (int r = 0; r < 8; ++r) {
                float p = __expf(sc[kt][r] - mnew[r]);
                rsum[r] += p;
                Ps[wave][(half * 8 + r) * LD + kt * 16 + l16] = (bf16)p;
            }
        }

        #pragma unroll
        for (int r = 0; r < 8; ++r) {
            float s = rsum[r];
            #pragma unroll
            for (int st = 1; st < 16; st <<= 1)
                s += __shfl_xor(s, st, 32);
            l8[r] = l8[r] * alpha[r] + s;
            m8[r] = mnew[r];
            #pragma unroll
            for (int ni = 0; ni < 4; ++ni) Oacc[ni][r] *= alpha[r];
        }

        // ---- O += P V (K-dim = keys, N = head dim) ----
        #pragma unroll
        for (int ks = 0; ks < 2; ++ks) {
            int kb = ks * 32 + half * 8;
            v8bf lo = *(const v8bf*)&Ps[wave][l16 * LD + kb];
            v8bf hi = *(const v8bf*)&Ps[wave][l16 * LD + kb + 16];
            v16bf ap;
            #pragma unroll
            for (int j = 0; j < 8; ++j) { ap[j] = lo[j]; ap[j + 8] = hi[j]; }
            #pragma unroll
            for (int ni = 0; ni < 4; ++ni) {
                int dcol = ni * 16 + l16;
                int kb2  = ks * 32 + half * 8;
                v8bf blo = *(const v8bf*)&Vt[buf][dcol * LD + kb2];
                v8bf bhi = *(const v8bf*)&Vt[buf][dcol * LD + kb2 + 16];
                v16bf bv;
                #pragma unroll
                for (int j = 0; j < 8; ++j) { bv[j] = blo[j]; bv[j + 8] = bhi[j]; }
                Oacc[ni] = __builtin_amdgcn_wmma_f32_16x16x32_bf16(
                    false, ap, false, bv, (short)0, Oacc[ni], false, false);
            }
        }

        async_wait_all();
        __syncthreads();
    }

    // ---- normalize + write ctx (bf16) ----
    #pragma unroll
    for (int ni = 0; ni < 4; ++ni)
        #pragma unroll
        for (int r = 0; r < 8; ++r) {
            int row = q0 + wave * 16 + half * 8 + r;
            int col = h * HEAD_DIM + ni * 16 + l16;
            float o = Oacc[ni][r] / l8[r];
            Ctx[((size_t)b * SEQ + row) * D_MODEL + col] = (bf16)o;
        }
}

// ---------------------------------------------------------------------------
extern "C" void kernel_launch(void* const* d_in, const int* in_sizes, int n_in,
                              void* d_out, int out_size, void* d_ws, size_t ws_size,
                              hipStream_t stream)
{
    (void)in_sizes; (void)n_in; (void)out_size; (void)ws_size;

    const float* k_in = (const float*)d_in[0];
    const float* q_in = (const float*)d_in[1];
    const float* v_in = (const float*)d_in[2];
    const float* w_k  = (const float*)d_in[3];
    const float* b_k  = (const float*)d_in[4];
    const float* w_q  = (const float*)d_in[5];
    const float* b_q  = (const float*)d_in[6];
    const float* w_v  = (const float*)d_in[7];
    const float* b_v  = (const float*)d_in[8];
    const float* w_o  = (const float*)d_in[9];
    const float* b_o  = (const float*)d_in[10];
    float* out = (float*)d_out;

    const size_t mat = (size_t)M_TOTAL * D_MODEL;   // 8M elements
    bf16* qp  = (bf16*)d_ws;
    bf16* kp  = qp + mat;
    bf16* vp  = kp + mat;
    bf16* ctx = vp + mat;

    dim3 gblk(256);
    dim3 ggrid(M_TOTAL / 128, D_MODEL / 128);       // 64 x 8

    const float qscale = 1.0f / (float)HEAD_DIM;    // fold scores/HEAD_DIM into Q

    gemm_bias_kernel<true, true><<<ggrid, gblk, 0, stream>>>(
        q_in, w_q, b_q, qp, M_TOTAL, D_MODEL, D_MODEL, qscale);
    gemm_bias_kernel<true, true><<<ggrid, gblk, 0, stream>>>(
        k_in, w_k, b_k, kp, M_TOTAL, D_MODEL, D_MODEL, 1.0f);
    gemm_bias_kernel<true, true><<<ggrid, gblk, 0, stream>>>(
        v_in, w_v, b_v, vp, M_TOTAL, D_MODEL, D_MODEL, 1.0f);

    dim3 agrid(SEQ / 128, NUM_HEADS, BATCH);        // 16 x 16 x 4
    attn_kernel<<<agrid, gblk, 0, stream>>>(qp, kp, vp, ctx);

    gemm_bias_kernel<false, false><<<ggrid, gblk, 0, stream>>>(
        ctx, w_o, b_o, out, M_TOTAL, D_MODEL, D_MODEL, 1.0f);
}
